// CircumpunctAttention_59287728554103
// MI455X (gfx1250) — matrix operand order repaired and checked by reading, not compiled, not run on hardware
//
#include <hip/hip_runtime.h>
#include <hip/hip_bf16.h>
#include <math.h>

#define D_MODEL 1024
#define N_HEADS 16
#define D_HEAD  64
#define SEQ_B   2
#define SEQ_T   2048
#define M_TOTAL (SEQ_B*SEQ_T)   // 4096 rows

typedef __attribute__((ext_vector_type(8)))  float          f32x8;
typedef __attribute__((ext_vector_type(16))) __bf16         bf16x16;
typedef __attribute__((ext_vector_type(8)))  unsigned short u16x8;
typedef __attribute__((ext_vector_type(16))) unsigned short u16x16;

union FragB { u16x16 u; bf16x16 b; };
union FragA { u16x16 u; bf16x16 b; u16x8 h[2]; };

__device__ __forceinline__ unsigned short f2bf(float f) {
  unsigned int u = __float_as_uint(f);
  u += 0x7FFFu + ((u >> 16) & 1u);           // round-to-nearest-even
  return (unsigned short)(u >> 16);
}

// ---------------- fp32 -> bf16 convert ----------------
__global__ void cvt_f32_bf16(const float* __restrict__ src,
                             unsigned short* __restrict__ dst, int n) {
  int i = blockIdx.x * blockDim.x + threadIdx.x;
  int stride = gridDim.x * blockDim.x;
  for (; i < n; i += stride) dst[i] = f2bf(src[i]);
}

// ---------------- GEMM:  Y[M,N] = X[M,K] @ W[N,K]^T  (bf16 in, f32 acc) ----
// Wave tile 32x64: 2 M-subtiles x 4 N-subtiles -> every B-frag load feeds two
// WMMAs. 8 waves/block as 4(M) x 2(N) -> 128x128 block tile.
template <bool OUT_BF16>
__global__ __launch_bounds__(256) void gemm_xwT(
    const unsigned short* __restrict__ X,   // M x K bf16
    const unsigned short* __restrict__ W,   // N x K bf16
    float* __restrict__ outF, unsigned short* __restrict__ outB,
    int M, int N, int K)
{
  const int lane = threadIdx.x & 31;
  const int wv   = threadIdx.x >> 5;
  const int l16  = lane & 15;
  const int sel  = lane >> 4;
  const int wm   = wv & 3;
  const int wn   = wv >> 2;
  const int m0 = blockIdx.y * 128 + wm * 32;
  const int n0 = blockIdx.x * 128 + wn * 64;

  f32x8 acc[2][4];
#pragma unroll
  for (int mt = 0; mt < 2; ++mt)
#pragma unroll
    for (int nt = 0; nt < 4; ++nt) acc[mt][nt] = 0.0f;

  const unsigned short* xrow0 = X + (size_t)(m0 + l16) * K;
  const unsigned short* xrow1 = X + (size_t)(m0 + 16 + l16) * K;
  for (int k = 0; k < K; k += 32) {
    FragA a0, a1;
    a0.h[0] = *(const u16x8*)(xrow0 + k + sel * 16);
    a0.h[1] = *(const u16x8*)(xrow0 + k + 8 + sel * 16);
    a1.h[0] = *(const u16x8*)(xrow1 + k + sel * 16);
    a1.h[1] = *(const u16x8*)(xrow1 + k + 8 + sel * 16);
#pragma unroll
    for (int nt = 0; nt < 4; ++nt) {
      const unsigned short* wrow =
          W + (size_t)(n0 + nt * 16 + l16) * K + k + sel * 16;
      FragB bf_; bf_.u = *(const u16x16*)wrow;
      acc[0][nt] = __builtin_amdgcn_wmma_f32_16x16x32_bf16(
          false, a0.b, false, bf_.b, (short)0, acc[0][nt], false, false);
      acc[1][nt] = __builtin_amdgcn_wmma_f32_16x16x32_bf16(
          false, a1.b, false, bf_.b, (short)0, acc[1][nt], false, false);
    }
  }
  // C layout: VGPR j -> row +j (lanes 0-15) / +j+8 (lanes 16-31), col = lane&15
#pragma unroll
  for (int mt = 0; mt < 2; ++mt) {
#pragma unroll
    for (int nt = 0; nt < 4; ++nt) {
      int c = n0 + nt * 16 + l16;
#pragma unroll
      for (int j = 0; j < 8; ++j) {
        int r = m0 + mt * 16 + j + 8 * sel;
        if (OUT_BF16) outB[(size_t)r * N + c] = f2bf(acc[mt][nt][j]);
        else          outF[(size_t)r * N + c] = acc[mt][nt][j];
      }
    }
  }
}

// ---------------- V transpose: vt[b][h][d][t] = v[b][t][h*64+d] ------------
__global__ void transposeV(const unsigned short* __restrict__ vb,
                           unsigned short* __restrict__ vt, int total) {
  int i = blockIdx.x * blockDim.x + threadIdx.x;
  if (i >= total) return;
  int t = i & (SEQ_T - 1);
  int d = (i >> 11) & 63;
  int h = (i >> 17) & 15;
  int b = i >> 21;
  vt[i] = vb[(size_t)(b * SEQ_T + t) * D_MODEL + h * D_HEAD + d];
}

// ---------------- Flash attention + rotation/valve epilogue ----------------
// 4 waves/block, each wave owns one (b,h,16-query tile). 64-key chunks:
// 8 score WMMAs + 8 PV WMMAs + 2 ones-WMMAs (row sums on the matrix pipe).
// Only the row-max reduction uses cross-lane shuffles.
__global__ __launch_bounds__(128) void flash_attn(
    const unsigned short* __restrict__ qb,
    const unsigned short* __restrict__ kb,
    const unsigned short* __restrict__ vt,   // per-head dh x T
    const float* __restrict__ beta,
    const float* __restrict__ ivv,
    const float* __restrict__ ovv,
    const float* __restrict__ chi,
    unsigned short* __restrict__ ao)
{
  __shared__ __align__(16) unsigned short ptile[4][16 * 64];  // 2KB per wave

  const int lane = threadIdx.x & 31;
  const int wv   = threadIdx.x >> 5;
  const int l16  = lane & 15;
  const int sel  = lane >> 4;

  const int QT   = SEQ_T / 16;            // 128 query tiles per (b,h)
  int widx = blockIdx.x * 4 + wv;
  int qt = widx % QT;
  int h  = (widx / QT) % N_HEADS;
  int b  = widx / (QT * N_HEADS);
  int q0 = qt * 16;

  const float inv_scale = 0.125f;         // 1/sqrt(dh * phi^0) = 1/8

  // Q fragments (16 queries x 64 dh = two K-steps), loaded once
  const unsigned short* qrow =
      qb + ((size_t)(b * SEQ_T) + q0 + l16) * D_MODEL + h * D_HEAD;
  FragA qf[2];
#pragma unroll
  for (int ks = 0; ks < 2; ++ks) {
    qf[ks].h[0] = *(const u16x8*)(qrow + ks * 32 + sel * 16);
    qf[ks].h[1] = *(const u16x8*)(qrow + ks * 32 + 8 + sel * 16);
  }

  // all-ones B fragment (bf16 1.0) for row-sum WMMAs
  FragB ones;
#pragma unroll
  for (int i = 0; i < 16; ++i) ones.u[i] = 0x3F80;

  f32x8 acc[4];
  f32x8 lacc;                              // row-sum accumulator
#pragma unroll
  for (int t = 0; t < 4; ++t) acc[t] = 0.0f;
  lacc = 0.0f;
  float mrun[8];
#pragma unroll
  for (int j = 0; j < 8; ++j) mrun[j] = -3.0e38f;

  const unsigned short* kbase = kb + (size_t)(b * SEQ_T) * D_MODEL + h * D_HEAD;
  const unsigned short* vbase =
      vt + (size_t)(b * N_HEADS + h) * D_HEAD * SEQ_T;
  unsigned short* myp = &ptile[wv][0];

  for (int c = 0; c < SEQ_T; c += 64) {
    // prefetch next chunk's K rows (cacheline covers the whole 128B head slice)
    if (c + 64 < SEQ_T) {
#pragma unroll
      for (int kt = 0; kt < 4; ++kt)
        __builtin_prefetch(kbase + (size_t)(c + 64 + kt * 16 + l16) * D_MODEL,
                           0, 1);
    }

    // ---- scores S = Q K^T for 64 keys (four 16-key N tiles) ----
    f32x8 s[4];
#pragma unroll
    for (int kt = 0; kt < 4; ++kt) s[kt] = 0.0f;
#pragma unroll
    for (int kt = 0; kt < 4; ++kt) {
      const unsigned short* krow =
          kbase + (size_t)(c + kt * 16 + l16) * D_MODEL;
#pragma unroll
      for (int ks = 0; ks < 2; ++ks) {
        FragB bf_; bf_.u = *(const u16x16*)(krow + ks * 32 + sel * 16);
        s[kt] = __builtin_amdgcn_wmma_f32_16x16x32_bf16(
            false, qf[ks].b, false, bf_.b, (short)0, s[kt], false, false);
      }
    }

    // ---- online softmax: new max, alpha, P = exp(S - m); P -> LDS ----
    float alpha[8];
#pragma unroll
    for (int j = 0; j < 8; ++j) {
      float a0 = s[0][j] * inv_scale;
      float a1 = s[1][j] * inv_scale;
      float a2 = s[2][j] * inv_scale;
      float a3 = s[3][j] * inv_scale;
      float cm = fmaxf(fmaxf(a0, a1), fmaxf(a2, a3));
      cm = fmaxf(cm, __shfl_xor(cm, 1, 32));
      cm = fmaxf(cm, __shfl_xor(cm, 2, 32));
      cm = fmaxf(cm, __shfl_xor(cm, 4, 32));
      cm = fmaxf(cm, __shfl_xor(cm, 8, 32));
      float mn = fmaxf(mrun[j], cm);
      alpha[j] = __expf(mrun[j] - mn);
      mrun[j]  = mn;
      int r = j + 8 * sel;
      myp[r * 64 + l16]      = f2bf(__expf(a0 - mn));
      myp[r * 64 + 16 + l16] = f2bf(__expf(a1 - mn));
      myp[r * 64 + 32 + l16] = f2bf(__expf(a2 - mn));
      myp[r * 64 + 48 + l16] = f2bf(__expf(a3 - mn));
    }
    // rescale running accumulators (output + row-sum) by alpha
#pragma unroll
    for (int dt = 0; dt < 4; ++dt)
#pragma unroll
      for (int j = 0; j < 8; ++j) acc[dt][j] *= alpha[j];
#pragma unroll
    for (int j = 0; j < 8; ++j) lacc[j] *= alpha[j];

    __builtin_amdgcn_wave_barrier();   // DS ops are in-order within a wave
    // re-read P as A-matrix fragments (16x64 -> two K-steps)
    FragA pf[2];
#pragma unroll
    for (int ks = 0; ks < 2; ++ks) {
      pf[ks].h[0] = *(const u16x8*)(myp + l16 * 64 + ks * 32 + sel * 16);
      pf[ks].h[1] = *(const u16x8*)(myp + l16 * 64 + ks * 32 + 8 + sel * 16);
    }
    __builtin_amdgcn_wave_barrier();

    // ---- acc += P @ V_chunk ; lacc += P @ ones (row sums, matrix pipe) ----
#pragma unroll
    for (int ks = 0; ks < 2; ++ks) {
#pragma unroll
      for (int dt = 0; dt < 4; ++dt) {
        const unsigned short* vrow =
            vbase + (size_t)(dt * 16 + l16) * SEQ_T + c + ks * 32 + sel * 16;
        FragB bf_; bf_.u = *(const u16x16*)vrow;
        acc[dt] = __builtin_amdgcn_wmma_f32_16x16x32_bf16(
            false, pf[ks].b, false, bf_.b, (short)0, acc[dt], false, false);
      }
      lacc = __builtin_amdgcn_wmma_f32_16x16x32_bf16(
          false, pf[ks].b, false, ones.b, (short)0, lacc, false, false);
    }
  }

  // ---- fused epilogue: /l, input/output valves, chi gate, half-dim rotation
  float bsig = 1.0f / (1.0f + __expf(-beta[h]));
  float ivh  = 1.0f / (1.0f + __expf(-ivv[h]));
  float ovh  = 1.0f / (1.0f + __expf(-ovv[h]));
  float gh   = tanhf(chi[h]);
  float ang  = 3.14159265358979323846f * bsig;
  float ca = __cosf(ang), sa = __sinf(ang);
  float gate = ivh * ovh * gh;

#pragma unroll
  for (int j = 0; j < 8; ++j) {
    int r = q0 + j + 8 * sel;
    float g_l = gate / lacc[j];
    size_t base = ((size_t)(b * SEQ_T) + r) * D_MODEL + h * D_HEAD;
#pragma unroll
    for (int dt = 0; dt < 2; ++dt) {         // d and d+32 pair up
      float re = acc[dt][j];
      float im = acc[dt + 2][j];
      float o_re = (re * ca - im * sa) * g_l;
      float o_im = (re * sa + im * ca) * g_l;
      ao[base + dt * 16 + l16]      = f2bf(o_re);
      ao[base + 32 + dt * 16 + l16] = f2bf(o_im);
    }
  }
}

// ---------------------------- host orchestration ---------------------------
extern "C" void kernel_launch(void* const* d_in, const int* in_sizes, int n_in,
                              void* d_out, int out_size, void* d_ws, size_t ws_size,
                              hipStream_t stream) {
  const float* x    = (const float*)d_in[0];
  const float* Wq   = (const float*)d_in[1];
  const float* Wk   = (const float*)d_in[2];
  const float* Wv   = (const float*)d_in[3];
  const float* We   = (const float*)d_in[4];
  const float* beta = (const float*)d_in[5];
  const float* iv   = (const float*)d_in[6];
  const float* ov   = (const float*)d_in[7];
  const float* chi  = (const float*)d_in[8];
  float* out = (float*)d_out;

  char* ws = (char*)d_ws;
  size_t off = 0;
  auto alloc = [&](size_t bytes) {
    void* p = ws + off;
    off += (bytes + 255) & ~(size_t)255;
    return p;
  };
  const size_t XB = (size_t)M_TOTAL * D_MODEL * 2;   // 8 MB bf16 buffers
  const size_t WB = (size_t)D_MODEL * D_MODEL * 2;   // 2 MB bf16 weights
  unsigned short* xb  = (unsigned short*)alloc(XB);
  unsigned short* wqb = (unsigned short*)alloc(WB);
  unsigned short* wkb = (unsigned short*)alloc(WB);
  unsigned short* wvb = (unsigned short*)alloc(WB);
  unsigned short* web = (unsigned short*)alloc(WB);
  unsigned short* qbv = (unsigned short*)alloc(XB);
  unsigned short* kbv = (unsigned short*)alloc(XB);
  unsigned short* vbv = (unsigned short*)alloc(XB);
  unsigned short* vtr = (unsigned short*)alloc(XB);
  unsigned short* aov = (unsigned short*)alloc(XB);

  const int nX = M_TOTAL * D_MODEL;          // 4M
  const int nW = D_MODEL * D_MODEL;          // 1M
  cvt_f32_bf16<<<1024, 256, 0, stream>>>(x,  xb,  nX);
  cvt_f32_bf16<<<256, 256, 0, stream>>>(Wq, wqb, nW);
  cvt_f32_bf16<<<256, 256, 0, stream>>>(Wk, wkb, nW);
  cvt_f32_bf16<<<256, 256, 0, stream>>>(Wv, wvb, nW);
  cvt_f32_bf16<<<256, 256, 0, stream>>>(We, web, nW);

  dim3 gg(D_MODEL / 128, M_TOTAL / 128);     // (8, 32)
  gemm_xwT<true><<<gg, 256, 0, stream>>>(xb, wqb, nullptr, qbv,
                                         M_TOTAL, D_MODEL, D_MODEL);
  gemm_xwT<true><<<gg, 256, 0, stream>>>(xb, wkb, nullptr, kbv,
                                         M_TOTAL, D_MODEL, D_MODEL);
  gemm_xwT<true><<<gg, 256, 0, stream>>>(xb, wvb, nullptr, vbv,
                                         M_TOTAL, D_MODEL, D_MODEL);

  transposeV<<<(nX + 255) / 256, 256, 0, stream>>>(vbv, vtr, nX);

  flash_attn<<<(SEQ_B * N_HEADS * (SEQ_T / 16)) / 4, 128, 0, stream>>>(
      qbv, kbv, vtr, beta, iv, ov, chi, aov);

  gemm_xwT<false><<<gg, 256, 0, stream>>>(aov, web, out, nullptr,
                                          M_TOTAL, D_MODEL, D_MODEL);
}
